// HFScatterMoEGatedMLP_33998961115696
// MI455X (gfx1250) — matrix operand order repaired and checked by reading, not compiled, not run on hardware
//
#include <hip/hip_runtime.h>

// ---------------------------------------------------------------------------
// Qwen2-MoE gated MLP for MI455X (gfx1250, wave32, v_wmma_f32_16x16x32_bf16).
// Round 3: double-buffered LDS tiles staged with CDNA5 async copies
// (global_load_async_to_lds_b128, ASYNCcnt-tracked, no VGPR round-trip) so
// global latency overlaps the WMMA sweep. Weights pre-converted to bf16 once
// per call (expert weights transposed so every GEMM B is k-contiguous).
// ---------------------------------------------------------------------------

#define N_TOK 2048
#define DMODEL 1024
#define NEXP 16
#define MOE_FF 512
#define SFF 2816

typedef __attribute__((ext_vector_type(16))) __bf16 v16bf;
typedef __attribute__((ext_vector_type(2)))  __bf16 v2bf;
typedef __attribute__((ext_vector_type(8)))  float  v8f;

#define LDS_STRIDE 40   // ushorts per LDS tile row (32 + 8 pad) = 80 B

__device__ inline unsigned short f2bf(float f) {
    unsigned u = __float_as_uint(f);
    return (unsigned short)((u + 0x7FFFu + ((u >> 16) & 1u)) >> 16);  // RNE
}

__device__ inline unsigned pack2bf(float a, float b) {
#if __has_builtin(__builtin_amdgcn_cvt_pk_bf16_f32)
    union { v2bf v; unsigned u; } c;
    c.v = __builtin_amdgcn_cvt_pk_bf16_f32(a, b);
    return c.u;
#else
    return (unsigned)f2bf(a) | ((unsigned)f2bf(b) << 16);
#endif
}

__device__ inline float silu_f(float g) {
    return g * __builtin_amdgcn_rcpf(1.f + __expf(-g));
}

union FragU { v16bf v; uint4 u[2]; };

// A fragment (16x32 bf16): lane = row M=lane&15; K-halves by lane>=16.
__device__ inline v16bf frag_a(const unsigned short* As, int mBase, int lane) {
    int row = mBase + (lane & 15);
    int kb  = (lane >> 4) * 8;
    const unsigned short* p = As + row * LDS_STRIDE + kb;
    FragU f;
    f.u[0] = *reinterpret_cast<const uint4*>(p);
    f.u[1] = *reinterpret_cast<const uint4*>(p + 16);
    return f.v;
}

// B fragment (32x16 bf16), Bs stored [n][k]: lane = col N=lane&15,
// K-halves by lane>=16 (same order as frag_a).
__device__ inline v16bf frag_b(const unsigned short* Bs, int nBase, int lane) {
    int col = nBase + (lane & 15);
    int kb  = (lane >> 4) * 8;
    const unsigned short* p = Bs + col * LDS_STRIDE + kb;
    FragU f;
    f.u[0] = *reinterpret_cast<const uint4*>(p);
    f.u[1] = *reinterpret_cast<const uint4*>(p + 16);
    return f.v;
}

__device__ inline v8f wmma_bf16(v16bf a, v16bf b, v8f c) {
    return __builtin_amdgcn_wmma_f32_16x16x32_bf16(false, a, false, b,
                                                   (short)0, c, false, false);
}

__device__ inline void wait_async_all() {
#if __has_builtin(__builtin_amdgcn_s_wait_asynccnt)
    __builtin_amdgcn_s_wait_asynccnt(0);
#else
    asm volatile("s_wait_asynccnt 0" ::: "memory");
#endif
}

// Stage a 128-row x 32-col bf16 tile into LDS with CDNA5 async copies:
// each lane issues global_load_async_to_lds_b128 (16 B), no VGPR data,
// tracked by ASYNCcnt. rowIdx (LDS) optionally gathers rows (neg -> 0).
__device__ inline void stage128_async(unsigned short* lds,
                                      const unsigned short* base, size_t ld,
                                      int k0, int tid, const int* rowIdx) {
#pragma unroll
    for (int r = 0; r < 2; ++r) {
        int idx = tid + r * 256;            // 512 uint4 slots
        int row = idx >> 2, ko = (idx & 3) * 8;
        int gr = row;
        if (rowIdx) { gr = rowIdx[row]; if (gr < 0) gr = 0; }
        const unsigned short* gp = base + (size_t)gr * ld + k0 + ko;
        unsigned lAddr = (unsigned)(size_t)(lds + row * LDS_STRIDE + ko);
        asm volatile("global_load_async_to_lds_b128 %0, %1, off"
                     :: "v"(lAddr), "v"(gp) : "memory");
    }
}

// ---------------------------------------------------------------------------
// Weight conversion: fp32 -> bf16, vectorized (float4 -> 2x packed bf16).
// ---------------------------------------------------------------------------
__global__ __launch_bounds__(256) void convert_flat_kernel(
    const float4* __restrict__ src, uint2* __restrict__ dst)
{
    size_t i = (size_t)blockIdx.x * 256 + threadIdx.x;
    float4 v = src[i];
    uint2 p;
    p.x = pack2bf(v.x, v.y);
    p.y = pack2bf(v.z, v.w);
    dst[i] = p;
}

// Transpose-convert: src [K][Nn] fp32 (Nn contiguous) -> dst [Nn][K] bf16.
// blockIdx.z batches (experts); batch stride = K*Nn for both. 64x64 tiles.
__global__ __launch_bounds__(256) void convert_tr_kernel(
    const float* __restrict__ src, unsigned short* __restrict__ dst,
    int K, int Nn)
{
    __shared__ alignas(16) unsigned short T[64][68];
    size_t boff = (size_t)blockIdx.z * K * Nn;
    src += boff;
    dst += boff;
    int k0 = blockIdx.x * 64, n0 = blockIdx.y * 64;
    int tid = threadIdx.x;
#pragma unroll
    for (int r = 0; r < 4; ++r) {               // read 64x64 floats, coalesced
        int idx = tid + r * 256;
        int kk = idx >> 4, nq = (idx & 15) * 4;
        float4 v = *reinterpret_cast<const float4*>(
            src + (size_t)(k0 + kk) * Nn + n0 + nq);
        uint2 p;
        p.x = pack2bf(v.x, v.y);
        p.y = pack2bf(v.z, v.w);
        *reinterpret_cast<uint2*>(&T[kk][nq]) = p;
    }
    __syncthreads();
#pragma unroll
    for (int r = 0; r < 4; ++r) {               // write transposed, coalesced
        int idx = tid + r * 256;
        int nn = idx >> 4, kq = (idx & 15) * 4;
        uint2 p;
        p.x = (unsigned)T[kq + 0][nn] | ((unsigned)T[kq + 1][nn] << 16);
        p.y = (unsigned)T[kq + 2][nn] | ((unsigned)T[kq + 3][nn] << 16);
        *reinterpret_cast<uint2*>(dst + (size_t)(n0 + nn) * K + k0 + kq) = p;
    }
}

// ---------------------------------------------------------------------------
// Router: one wave per token; logits, softmax, top-2 renorm, sigmoid gate.
// ---------------------------------------------------------------------------
__global__ __launch_bounds__(256) void router_kernel(
    const float* __restrict__ x, const float* __restrict__ gate_w,
    const float* __restrict__ sgw, int* __restrict__ sel,
    float* __restrict__ rw, float* __restrict__ sgate)
{
    int lane = threadIdx.x & 31;
    int n = blockIdx.x * 8 + (threadIdx.x >> 5);
    const float* xr = x + (size_t)n * DMODEL;

    float acc[NEXP];
    float sacc = 0.f;
#pragma unroll
    for (int e = 0; e < NEXP; ++e) acc[e] = 0.f;

    for (int d = lane; d < DMODEL; d += 32) {
        float xv = xr[d];
#pragma unroll
        for (int e = 0; e < NEXP; ++e) acc[e] += xv * gate_w[e * DMODEL + d];
        sacc += xv * sgw[d];
    }
#pragma unroll
    for (int off = 16; off >= 1; off >>= 1) {
#pragma unroll
        for (int e = 0; e < NEXP; ++e) acc[e] += __shfl_xor(acc[e], off, 32);
        sacc += __shfl_xor(sacc, off, 32);
    }
    if (lane == 0) {
        float mx = acc[0];
#pragma unroll
        for (int e = 1; e < NEXP; ++e) mx = fmaxf(mx, acc[e]);
        float p[NEXP], s = 0.f;
#pragma unroll
        for (int e = 0; e < NEXP; ++e) { p[e] = __expf(acc[e] - mx); s += p[e]; }
        float inv_s = 1.f / s;
#pragma unroll
        for (int e = 0; e < NEXP; ++e) p[e] *= inv_s;
        int i0 = 0;
        for (int e = 1; e < NEXP; ++e) if (p[e] > p[i0]) i0 = e;
        float p0 = p[i0]; p[i0] = -1.f;
        int i1 = 0;
        for (int e = 1; e < NEXP; ++e) if (p[e] > p[i1]) i1 = e;
        float p1 = p[i1];
        float inv = 1.f / (p0 + p1);
        sel[n * 2 + 0] = i0;  sel[n * 2 + 1] = i1;
        rw[n * 2 + 0] = p0 * inv;  rw[n * 2 + 1] = p1 * inv;
        sgate[n] = 1.f / (1.f + __expf(-sacc));
    }
}

__global__ void zero_cnt_kernel(int* cnt) {
    if (threadIdx.x < NEXP) cnt[threadIdx.x] = 0;
}

__global__ __launch_bounds__(256) void scatter_kernel(
    const int* __restrict__ sel, const float* __restrict__ rw,
    int* __restrict__ cnt, int* __restrict__ btok, int* __restrict__ bdst,
    float* __restrict__ bw)
{
    int n = blockIdx.x * 256 + threadIdx.x;
#pragma unroll
    for (int k = 0; k < 2; ++k) {
        int e = sel[n * 2 + k];
        int pos = atomicAdd(&cnt[e], 1);
        int slot = e * N_TOK + pos;
        btok[slot] = n;
        bdst[slot] = k * N_TOK + n;
        bw[slot]   = rw[n * 2 + k];
    }
}

// ---------------------------------------------------------------------------
// Shared up: h_shared = silu(x@Wg^T) * (x@Wu^T); bf16 operands, dual-B,
// double-buffered async staging.
// ---------------------------------------------------------------------------
__global__ __launch_bounds__(256) void shared_up_kernel(
    const unsigned short* __restrict__ xb, const unsigned short* __restrict__ wg,
    const unsigned short* __restrict__ wu, unsigned short* __restrict__ h_out)
{
    __shared__ alignas(16) unsigned short As[2][128 * LDS_STRIDE];
    __shared__ alignas(16) unsigned short Bg[2][128 * LDS_STRIDE];
    __shared__ alignas(16) unsigned short Bu[2][128 * LDS_STRIDE];

    int tid = threadIdx.x, lane = tid & 31, wave = tid >> 5;
    int wm = (wave >> 2) * 64, wn = (wave & 3) * 32;
    int m0 = blockIdx.y * 128, n0 = blockIdx.x * 128;
    const unsigned short* aBase = xb + (size_t)m0 * DMODEL;
    const unsigned short* gBase = wg + (size_t)n0 * DMODEL;
    const unsigned short* uBase = wu + (size_t)n0 * DMODEL;

    v8f ag[4][2] = {}, au[4][2] = {};
    const int KT = DMODEL / 32;

    stage128_async(As[0], aBase, DMODEL, 0, tid, nullptr);
    stage128_async(Bg[0], gBase, DMODEL, 0, tid, nullptr);
    stage128_async(Bu[0], uBase, DMODEL, 0, tid, nullptr);

    for (int kt = 0; kt < KT; ++kt) {
        wait_async_all();
        __syncthreads();                       // buf[kt&1] complete block-wide
        if (kt + 1 < KT) {
            int nb = (kt + 1) & 1, nk = (kt + 1) * 32;
            stage128_async(As[nb], aBase, DMODEL, nk, tid, nullptr);
            stage128_async(Bg[nb], gBase, DMODEL, nk, tid, nullptr);
            stage128_async(Bu[nb], uBase, DMODEL, nk, tid, nullptr);
        }
        int b = kt & 1;
#pragma unroll
        for (int mt = 0; mt < 4; ++mt) {
            v16bf a = frag_a(As[b], wm + mt * 16, lane);
#pragma unroll
            for (int nt = 0; nt < 2; ++nt) {
                ag[mt][nt] = wmma_bf16(a, frag_b(Bg[b], wn + nt * 16, lane), ag[mt][nt]);
                au[mt][nt] = wmma_bf16(a, frag_b(Bu[b], wn + nt * 16, lane), au[mt][nt]);
            }
        }
        __syncthreads();                       // done reading buf[kt&1]
    }

    int rlo = lane & 15, hi = lane >> 4;
#pragma unroll
    for (int mt = 0; mt < 4; ++mt)
#pragma unroll
        for (int nt = 0; nt < 2; ++nt) {
            int n = n0 + wn + nt * 16 + rlo;
#pragma unroll
            for (int i = 0; i < 8; ++i) {
                int m = m0 + wm + mt * 16 + i + 8 * hi;
                float h = silu_f(ag[mt][nt][i]) * au[mt][nt][i];
                h_out[(size_t)m * SFF + n] = f2bf(h);
            }
        }
}

// ---------------------------------------------------------------------------
// Shared down: shared_out = (h_shared @ Wd^T) * sigmoid-gate.
// ---------------------------------------------------------------------------
__global__ __launch_bounds__(256) void shared_down_kernel(
    const unsigned short* __restrict__ h, const unsigned short* __restrict__ wd,
    const float* __restrict__ sgate, float* __restrict__ shout)
{
    __shared__ alignas(16) unsigned short As[2][128 * LDS_STRIDE];
    __shared__ alignas(16) unsigned short Bs[2][128 * LDS_STRIDE];

    int tid = threadIdx.x, lane = tid & 31, wave = tid >> 5;
    int wm = (wave >> 2) * 64, wn = (wave & 3) * 32;
    int m0 = blockIdx.y * 128, n0 = blockIdx.x * 128;
    const unsigned short* aBase = h + (size_t)m0 * SFF;
    const unsigned short* bBase = wd + (size_t)n0 * SFF;

    v8f acc[4][2] = {};
    const int KT = SFF / 32;

    stage128_async(As[0], aBase, SFF, 0, tid, nullptr);
    stage128_async(Bs[0], bBase, SFF, 0, tid, nullptr);

    for (int kt = 0; kt < KT; ++kt) {
        wait_async_all();
        __syncthreads();
        if (kt + 1 < KT) {
            int nb = (kt + 1) & 1, nk = (kt + 1) * 32;
            stage128_async(As[nb], aBase, SFF, nk, tid, nullptr);
            stage128_async(Bs[nb], bBase, SFF, nk, tid, nullptr);
        }
        int b = kt & 1;
#pragma unroll
        for (int mt = 0; mt < 4; ++mt) {
            v16bf a = frag_a(As[b], wm + mt * 16, lane);
#pragma unroll
            for (int nt = 0; nt < 2; ++nt)
                acc[mt][nt] = wmma_bf16(a, frag_b(Bs[b], wn + nt * 16, lane), acc[mt][nt]);
        }
        __syncthreads();
    }

    int rlo = lane & 15, hi = lane >> 4;
#pragma unroll
    for (int mt = 0; mt < 4; ++mt)
#pragma unroll
        for (int nt = 0; nt < 2; ++nt) {
            int n = n0 + wn + nt * 16 + rlo;
#pragma unroll
            for (int i = 0; i < 8; ++i) {
                int m = m0 + wm + mt * 16 + i + 8 * hi;
                shout[(size_t)m * DMODEL + n] = acc[mt][nt][i] * sgate[m];
            }
        }
}

// ---------------------------------------------------------------------------
// Expert up: gather bucket rows of x(bf16), fused gate|up GEMM vs transposed
// bf16 weights wgut[e][f][d], SwiGLU -> h_moe[dst].
// ---------------------------------------------------------------------------
__global__ __launch_bounds__(256) void moe_up_kernel(
    const unsigned short* __restrict__ xb, const unsigned short* __restrict__ wgut,
    const int* __restrict__ cnt, const int* __restrict__ btok,
    const int* __restrict__ bdst, unsigned short* __restrict__ h_moe)
{
    int e = blockIdx.z;
    int count = cnt[e];
    int m0 = blockIdx.y * 128;
    if (m0 >= count) return;

    __shared__ alignas(16) unsigned short As[2][128 * LDS_STRIDE];
    __shared__ alignas(16) unsigned short Bg[2][128 * LDS_STRIDE];
    __shared__ alignas(16) unsigned short Bu[2][128 * LDS_STRIDE];
    __shared__ int tokL[128];
    __shared__ int dstL[128];

    int tid = threadIdx.x, lane = tid & 31, wave = tid >> 5;
    int wm = (wave >> 2) * 64, wn = (wave & 3) * 32;
    int n0 = blockIdx.x * 128;
    const unsigned short* wbase = wgut + (size_t)e * DMODEL * (2 * MOE_FF);
    const unsigned short* gBase = wbase + (size_t)n0 * DMODEL;
    const unsigned short* uBase = wbase + (size_t)(n0 + MOE_FF) * DMODEL;

    if (tid < 128) {
        int r = m0 + tid;
        tokL[tid] = (r < count) ? btok[e * N_TOK + r] : 0;
        dstL[tid] = (r < count) ? bdst[e * N_TOK + r] : -1;
    }
    __syncthreads();

    v8f ag[4][2] = {}, au[4][2] = {};
    const int KT = DMODEL / 32;

    stage128_async(As[0], xb, DMODEL, 0, tid, tokL);
    stage128_async(Bg[0], gBase, DMODEL, 0, tid, nullptr);
    stage128_async(Bu[0], uBase, DMODEL, 0, tid, nullptr);

    for (int kt = 0; kt < KT; ++kt) {
        wait_async_all();
        __syncthreads();
        if (kt + 1 < KT) {
            int nb = (kt + 1) & 1, nk = (kt + 1) * 32;
            stage128_async(As[nb], xb, DMODEL, nk, tid, tokL);
            stage128_async(Bg[nb], gBase, DMODEL, nk, tid, nullptr);
            stage128_async(Bu[nb], uBase, DMODEL, nk, tid, nullptr);
        }
        int b = kt & 1;
#pragma unroll
        for (int mt = 0; mt < 4; ++mt) {
            v16bf a = frag_a(As[b], wm + mt * 16, lane);
#pragma unroll
            for (int nt = 0; nt < 2; ++nt) {
                ag[mt][nt] = wmma_bf16(a, frag_b(Bg[b], wn + nt * 16, lane), ag[mt][nt]);
                au[mt][nt] = wmma_bf16(a, frag_b(Bu[b], wn + nt * 16, lane), au[mt][nt]);
            }
        }
        __syncthreads();
    }

    int rlo = lane & 15, hi = lane >> 4;
#pragma unroll
    for (int mt = 0; mt < 4; ++mt)
#pragma unroll
        for (int nt = 0; nt < 2; ++nt) {
            int n = n0 + wn + nt * 16 + rlo;
#pragma unroll
            for (int i = 0; i < 8; ++i) {
                int ml = wm + mt * 16 + i + 8 * hi;
                int dst = dstL[ml];
                if (dst >= 0) {
                    float hv = silu_f(ag[mt][nt][i]) * au[mt][nt][i];
                    h_moe[(size_t)dst * MOE_FF + n] = f2bf(hv);
                }
            }
        }
}

// ---------------------------------------------------------------------------
// Expert down: outk[dst] = rw * (h_moe[dst] @ w_down[e]), B from transposed
// bf16 wdnt[e][d][f].
// ---------------------------------------------------------------------------
__global__ __launch_bounds__(256) void moe_down_kernel(
    const unsigned short* __restrict__ h_moe, const unsigned short* __restrict__ wdnt,
    const int* __restrict__ cnt, const int* __restrict__ bdst,
    const float* __restrict__ bw, float* __restrict__ outk)
{
    int e = blockIdx.z;
    int count = cnt[e];
    int m0 = blockIdx.y * 128;
    if (m0 >= count) return;

    __shared__ alignas(16) unsigned short As[2][128 * LDS_STRIDE];
    __shared__ alignas(16) unsigned short Bs[2][128 * LDS_STRIDE];
    __shared__ int dstL[128];
    __shared__ float wL[128];

    int tid = threadIdx.x, lane = tid & 31, wave = tid >> 5;
    int wm = (wave >> 2) * 64, wn = (wave & 3) * 32;
    int n0 = blockIdx.x * 128;
    const unsigned short* bBase = wdnt + (size_t)e * DMODEL * MOE_FF
                                       + (size_t)n0 * MOE_FF;

    if (tid < 128) {
        int r = m0 + tid;
        dstL[tid] = (r < count) ? bdst[e * N_TOK + r] : -1;
        wL[tid]   = (r < count) ? bw[e * N_TOK + r] : 0.f;
    }
    __syncthreads();

    v8f acc[4][2] = {};
    const int KT = MOE_FF / 32;

    stage128_async(As[0], h_moe, MOE_FF, 0, tid, dstL);
    stage128_async(Bs[0], bBase, MOE_FF, 0, tid, nullptr);

    for (int kt = 0; kt < KT; ++kt) {
        wait_async_all();
        __syncthreads();
        if (kt + 1 < KT) {
            int nb = (kt + 1) & 1, nk = (kt + 1) * 32;
            stage128_async(As[nb], h_moe, MOE_FF, nk, tid, dstL);
            stage128_async(Bs[nb], bBase, MOE_FF, nk, tid, nullptr);
        }
        int b = kt & 1;
#pragma unroll
        for (int mt = 0; mt < 4; ++mt) {
            v16bf a = frag_a(As[b], wm + mt * 16, lane);
#pragma unroll
            for (int nt = 0; nt < 2; ++nt)
                acc[mt][nt] = wmma_bf16(a, frag_b(Bs[b], wn + nt * 16, lane), acc[mt][nt]);
        }
        __syncthreads();
    }

    int rlo = lane & 15, hi = lane >> 4;
#pragma unroll
    for (int mt = 0; mt < 4; ++mt)
#pragma unroll
        for (int nt = 0; nt < 2; ++nt) {
            int n = n0 + wn + nt * 16 + rlo;
#pragma unroll
            for (int i = 0; i < 8; ++i) {
                int ml = wm + mt * 16 + i + 8 * hi;
                int dst = dstL[ml];
                if (dst >= 0)
                    outk[(size_t)dst * DMODEL + n] = wL[ml] * acc[mt][nt][i];
            }
        }
}

// out = shared_out + outk(k=0) + outk(k=1), float4-vectorized
__global__ __launch_bounds__(256) void combine_kernel(
    const float4* __restrict__ sh, const float4* __restrict__ ok,
    float4* __restrict__ out)
{
    size_t i = (size_t)blockIdx.x * 256 + threadIdx.x;
    float4 a = sh[i];
    float4 b = ok[i];
    float4 c = ok[(size_t)(N_TOK * DMODEL / 4) + i];
    float4 r;
    r.x = a.x + b.x + c.x;
    r.y = a.y + b.y + c.y;
    r.z = a.z + b.z + c.z;
    r.w = a.w + b.w + c.w;
    out[i] = r;
}

// ---------------------------------------------------------------------------
extern "C" void kernel_launch(void* const* d_in, const int* in_sizes, int n_in,
                              void* d_out, int out_size, void* d_ws, size_t ws_size,
                              hipStream_t stream)
{
    (void)in_sizes; (void)n_in; (void)out_size; (void)ws_size;
    const float* x   = (const float*)d_in[0];
    const float* gw  = (const float*)d_in[1];
    const float* wgu = (const float*)d_in[2];
    const float* wdn = (const float*)d_in[3];
    const float* swg = (const float*)d_in[4];
    const float* swu = (const float*)d_in[5];
    const float* swd = (const float*)d_in[6];
    const float* sgw = (const float*)d_in[7];
    float* out = (float*)d_out;

    char* p = (char*)d_ws;
    auto alloc = [&](size_t bytes) -> char* {
        char* r = p;
        p += (bytes + 255) & ~(size_t)255;
        return r;
    };
    int*   cnt  = (int*)  alloc(NEXP * sizeof(int));
    int*   sel  = (int*)  alloc((size_t)N_TOK * 2 * sizeof(int));
    float* rw   = (float*)alloc((size_t)N_TOK * 2 * sizeof(float));
    float* sg   = (float*)alloc((size_t)N_TOK * sizeof(float));
    int*   btok = (int*)  alloc((size_t)NEXP * N_TOK * sizeof(int));
    int*   bdst = (int*)  alloc((size_t)NEXP * N_TOK * sizeof(int));
    float* bw   = (float*)alloc((size_t)NEXP * N_TOK * sizeof(float));
    unsigned short* xbf  = (unsigned short*)alloc((size_t)N_TOK * DMODEL * 2);
    unsigned short* wgbf = (unsigned short*)alloc((size_t)SFF * DMODEL * 2);
    unsigned short* wubf = (unsigned short*)alloc((size_t)SFF * DMODEL * 2);
    unsigned short* wdbf = (unsigned short*)alloc((size_t)DMODEL * SFF * 2);
    unsigned short* wgut = (unsigned short*)alloc((size_t)NEXP * DMODEL * 2 * MOE_FF * 2);
    unsigned short* wdnt = (unsigned short*)alloc((size_t)NEXP * MOE_FF * DMODEL * 2);
    unsigned short* hsh  = (unsigned short*)alloc((size_t)N_TOK * SFF * 2);
    unsigned short* hmoe = (unsigned short*)alloc((size_t)2 * N_TOK * MOE_FF * 2);
    float* shout = (float*)alloc((size_t)N_TOK * DMODEL * sizeof(float));
    float* outk  = (float*)alloc((size_t)2 * N_TOK * DMODEL * sizeof(float));

    // --- one-shot fp32 -> bf16 conversion of activations + all weights ---
    convert_flat_kernel<<<(N_TOK * DMODEL) / 1024, 256, 0, stream>>>(
        (const float4*)x, (uint2*)xbf);
    convert_flat_kernel<<<(SFF * DMODEL) / 1024, 256, 0, stream>>>(
        (const float4*)swg, (uint2*)wgbf);
    convert_flat_kernel<<<(SFF * DMODEL) / 1024, 256, 0, stream>>>(
        (const float4*)swu, (uint2*)wubf);
    convert_flat_kernel<<<(DMODEL * SFF) / 1024, 256, 0, stream>>>(
        (const float4*)swd, (uint2*)wdbf);
    // w_gate_up [e][d][f] -> [e][f][d];  w_down [e][f][d] -> [e][d][f]
    convert_tr_kernel<<<dim3(DMODEL / 64, (2 * MOE_FF) / 64, NEXP), 256, 0, stream>>>(
        wgu, wgut, DMODEL, 2 * MOE_FF);
    convert_tr_kernel<<<dim3(MOE_FF / 64, DMODEL / 64, NEXP), 256, 0, stream>>>(
        wdn, wdnt, MOE_FF, DMODEL);

    // --- routing ---
    router_kernel<<<N_TOK / 8, 256, 0, stream>>>(x, gw, sgw, sel, rw, sg);
    zero_cnt_kernel<<<1, 32, 0, stream>>>(cnt);
    scatter_kernel<<<N_TOK / 256, 256, 0, stream>>>(sel, rw, cnt, btok, bdst, bw);

    // --- shared expert ---
    shared_up_kernel<<<dim3(SFF / 128, N_TOK / 128), 256, 0, stream>>>(
        xbf, wgbf, wubf, hsh);
    shared_down_kernel<<<dim3(DMODEL / 128, N_TOK / 128), 256, 0, stream>>>(
        hsh, wdbf, sg, shout);

    // --- routed experts (top-2 dispatch) ---
    moe_up_kernel<<<dim3(MOE_FF / 128, N_TOK / 128, NEXP), 256, 0, stream>>>(
        xbf, wgut, cnt, btok, bdst, hmoe);
    moe_down_kernel<<<dim3(DMODEL / 128, N_TOK / 128, NEXP), 256, 0, stream>>>(
        hmoe, wdnt, cnt, bdst, bw, outk);

    combine_kernel<<<(N_TOK * DMODEL) / 1024, 256, 0, stream>>>(
        (const float4*)shout, (const float4*)outk, (float4*)out);
}
